// CLIPAttention_64604898066919
// MI455X (gfx1250) — matrix operand order repaired and checked by reading, not compiled, not run on hardware
//
#include <hip/hip_runtime.h>
#include <hip/hip_bf16.h>
#include <math.h>

typedef __attribute__((ext_vector_type(16))) _Float16 v16h;
typedef __attribute__((ext_vector_type(8)))  float    v8f;

#define WMMA_F32_F16(a, b, c) \
  __builtin_amdgcn_wmma_f32_16x16x32_f16(false, (a), false, (b), (short)0, (c), false, false)

static constexpr int B_ = 16, T_ = 1024, E_ = 1024, H_ = 16, DH_ = 64;
static constexpr int BT_ = B_ * T_;

// ---- A operand (16-bit, 16x32, MxK): lane = M (both halves), kh = lane>>4.
// VGPR j: j<4 -> K = 2j (+8 if kh); j>=4 -> K = 16+2(j-4) (+8 if kh).  (ISA 7.12.2)
__device__ __forceinline__ v16h load_a32k(const _Float16* rowBase, int kh) {
  union { v16h v; unsigned u[8]; } r;
  const unsigned* p = (const unsigned*)rowBase;  // 4B-aligned row start
#pragma unroll
  for (int j = 0; j < 8; ++j) {
    int kst = ((j < 4) ? (2 * j) : (16 + 2 * (j - 4))) + 8 * kh;  // even
    r.u[j] = p[kst >> 1];
  }
  return r.v;
}

// ---- B operand (16-bit, 32x16, KxN): lane = N (l&15), element e<8 -> K = e + 8*(lane>>4); e>=8 -> K = 16 + (e-8) + 8*(lane>>4).
__device__ __forceinline__ v16h load_b32k(const _Float16* colBase, int kh) {
  union { v16h v; uint2 u[4]; } r;
  const uint2* p = (const uint2*)(colBase + 8 * kh);  // 8B-aligned
#pragma unroll
  for (int j = 0; j < 4; ++j) r.u[j] = p[(j & 1) + (j >> 1) * 4];
  return r.v;
}

// ---- DPP16 row_ror rotation (stays inside 16-lane row => inside wave halves) ----
template <int N>
__device__ __forceinline__ float dpp_ror(float x) {
  int r = __builtin_amdgcn_update_dpp(0, __builtin_bit_cast(int, x),
                                      0x120 | N, 0xf, 0xf, true);
  return __builtin_bit_cast(float, r);
}
__device__ __forceinline__ float rmax16(float x) {   // max over 16-lane row, all lanes
  x = fmaxf(x, dpp_ror<1>(x));
  x = fmaxf(x, dpp_ror<2>(x));
  x = fmaxf(x, dpp_ror<4>(x));
  x = fmaxf(x, dpp_ror<8>(x));
  return x;
}

// ---------------- fp32 -> fp16 convert ----------------
__global__ void cvt_f32_f16(const float* __restrict__ in, _Float16* __restrict__ out, int n) {
  int i = blockIdx.x * blockDim.x + threadIdx.x;
  if (i < n) out[i] = (_Float16)in[i];
}

// ---------------- NT GEMM: C[m,n] = sum_k A[m,k]*W[n,k] + bias[n] ----------------
// WG tile 128x128, 8 waves (4M x 2N); software-pipelined: next global slab is
// staged in registers while the current LDS slab feeds the WMMAs.
template <bool F32OUT>
__global__ void __launch_bounds__(256, 1)
gemm_nt(const _Float16* __restrict__ A,   // M x K row-major
        const _Float16* __restrict__ W,   // N x K row-major
        const float* __restrict__ bias,   // N
        void* __restrict__ outp, int M, int N, int K) {
  constexpr int LDK = 40;  // 32 + pad, 80B stride
  __shared__ __align__(16) _Float16 lA[128][LDK];
  __shared__ __align__(16) _Float16 lB[128][LDK];

  const int tid = threadIdx.x, lane = tid & 31, wave = tid >> 5;
  const int wm = wave >> 1, wn = wave & 1;
  const int nb = N / 128;
  const int bm = blockIdx.x / nb, bn = blockIdx.x % nb;
  const int kh = lane >> 4, ln = lane & 15;

  v8f acc[2][4];
#pragma unroll
  for (int i = 0; i < 2; ++i)
#pragma unroll
    for (int j = 0; j < 4; ++j) acc[i][j] = (v8f){};

  const _Float16* Ab = A + (size_t)(bm * 128) * K;
  const _Float16* Wb = W + (size_t)(bn * 128) * K;

  uint2 stA[4], stB[4];
  const int srow = tid >> 3, skq = (tid & 7) * 4;     // this thread's 4 slab quads
  auto load_slab = [&](int k0) {
#pragma unroll
    for (int i = 0; i < 4; ++i) {
      int row = srow + i * 32;
      stA[i] = *(const uint2*)&Ab[(size_t)row * K + k0 + skq];
      stB[i] = *(const uint2*)&Wb[(size_t)row * K + k0 + skq];
    }
  };
  auto store_slab = [&]() {
#pragma unroll
    for (int i = 0; i < 4; ++i) {
      int row = srow + i * 32;
      *(uint2*)&lA[row][skq] = stA[i];
      *(uint2*)&lB[row][skq] = stB[i];
    }
  };

  load_slab(0);
  store_slab();
  __syncthreads();

  for (int k0 = 0; k0 < K; k0 += 32) {
    const bool more = (k0 + 32 < K);
    if (more) load_slab(k0 + 32);     // global fetch overlaps WMMA below

    v16h aop[2], bop[4];
#pragma unroll
    for (int mi = 0; mi < 2; ++mi) aop[mi] = load_a32k(&lA[wm * 32 + mi * 16 + ln][0], kh);
#pragma unroll
    for (int ni = 0; ni < 4; ++ni) bop[ni] = load_b32k(&lB[wn * 64 + ni * 16 + ln][0], kh);
#pragma unroll
    for (int ni = 0; ni < 4; ++ni)
#pragma unroll
      for (int mi = 0; mi < 2; ++mi) acc[mi][ni] = WMMA_F32_F16(aop[mi], bop[ni], acc[mi][ni]);

    if (more) {
      __syncthreads();
      store_slab();
      __syncthreads();
    }
  }

#pragma unroll
  for (int ni = 0; ni < 4; ++ni) {
    int gn = bn * 128 + wn * 64 + ni * 16 + ln;
    float bv = bias[gn];
#pragma unroll
    for (int mi = 0; mi < 2; ++mi)
#pragma unroll
      for (int r = 0; r < 8; ++r) {
        int gm = bm * 128 + wm * 32 + mi * 16 + r + 8 * kh;
        float v = acc[mi][ni][r] + bv;
        if constexpr (F32OUT) ((float*)outp)[(size_t)gm * N + gn] = v;
        else                  ((_Float16*)outp)[(size_t)gm * N + gn] = (_Float16)v;
      }
  }
}

// ---------------- 2-D RoPE on Q and K (in place), per channel pair ----------------
__global__ void rope2d(_Float16* __restrict__ Q, _Float16* __restrict__ Kb,
                       const int* __restrict__ pos) {
  int i = blockIdx.x * blockDim.x + threadIdx.x;  // BT * 512 pairs
  if (i >= BT_ * (E_ / 2)) return;
  int c = i & 511, bt = i >> 9;
  int h = c >> 5, j = c & 31;
  int whalf = j >> 4, idx = j & 15;                 // quarter = 16
  float p = (float)pos[bt * 2 + whalf];
  float inv = __powf(10000.0f, -(float)idx / 8.0f); // theta^(-2*idx/quarter)
  float sn, cs;
  __sincosf(p * inv, &sn, &cs);
  size_t base = (size_t)bt * E_ + h * 64 + whalf * 32 + 2 * idx;
  float q0 = (float)Q[base], q1 = (float)Q[base + 1];
  Q[base]     = (_Float16)(q0 * cs - q1 * sn);
  Q[base + 1] = (_Float16)(q1 * cs + q0 * sn);
  float k0 = (float)Kb[base], k1 = (float)Kb[base + 1];
  Kb[base]     = (_Float16)(k0 * cs - k1 * sn);
  Kb[base + 1] = (_Float16)(k1 * cs + k0 * sn);
}

// ---------------- Flash attention: WG = (b,h, 128 q rows), wave = 16 q rows ----------------
__global__ void __launch_bounds__(256, 1)
attn_flash(const _Float16* __restrict__ Q, const _Float16* __restrict__ Kb,
           const _Float16* __restrict__ V, _Float16* __restrict__ O) {
  __shared__ __align__(16) _Float16 lK[32][72];        // [kcol][d]   (B op of S)
  __shared__ __align__(16) _Float16 lV[64][40];        // [d][kcol]   (B op of PV)
  __shared__ __align__(16) _Float16 lP[8][16][40];     // per-wave P re-layout

  const int tid = threadIdx.x, lane = tid & 31, wave = tid >> 5;
  const int kh = lane >> 4, ln = lane & 15;
  const int blk = blockIdx.x;
  const int qt = blk & 7, h = (blk >> 3) & 15, b = blk >> 7;
  const int t0 = qt * 128 + wave * 16;
  const int hbase = h * 64;

  const size_t rowQ = ((size_t)b * T_ + t0 + ln) * E_ + hbase;
  const v16h aq0 = load_a32k(Q + rowQ, kh);        // d 0..31
  const v16h aq1 = load_a32k(Q + rowQ + 32, kh);   // d 32..63

  v16h ones;                                       // all-ones B: row-sum via WMMA
#pragma unroll
  for (int i = 0; i < 16; ++i) ones[i] = (_Float16)1.0f;

  v8f o0 = {}, o1 = {}, o2 = {}, o3 = {}, osum = {};
  float mrow[8];
#pragma unroll
  for (int r = 0; r < 8; ++r) mrow[r] = -__builtin_inff();
  const float scale = 0.125f;  // DH^-0.5

  for (int k0 = 0; k0 < T_; k0 += 32) {
    __syncthreads();
#pragma unroll
    for (int i = 0; i < 2; ++i) {                    // K chunk: 512 uint2
      int e = tid + i * 256;
      int kc = e >> 4, dq = (e & 15) * 4;
      *(uint2*)&lK[kc][dq] = *(const uint2*)&Kb[((size_t)b * T_ + k0 + kc) * E_ + hbase + dq];
    }
#pragma unroll
    for (int i = 0; i < 8; ++i) {                    // V chunk, transposed into LDS
      int e = tid + i * 256;
      int kc = e >> 6, d = e & 63;
      lV[d][kc] = V[((size_t)b * T_ + k0 + kc) * E_ + hbase + d];
    }
    __syncthreads();
    if (k0 + 32 < T_) {                              // prefetch next K/V chunk
      int kc = tid >> 3;                             // 0..31
      __builtin_prefetch(&Kb[((size_t)b * T_ + k0 + 32 + kc) * E_ + hbase], 0, 1);
      __builtin_prefetch(&V [((size_t)b * T_ + k0 + 32 + kc) * E_ + hbase], 0, 1);
    }

    // S = Q K^T for 32 k-columns (all B operands hoisted to overlap DS latency)
    v16h b00 = load_b32k(&lK[ln][0], kh);
    v16h b10 = load_b32k(&lK[16 + ln][0], kh);
    v16h b01 = load_b32k(&lK[ln][32], kh);
    v16h b11 = load_b32k(&lK[16 + ln][32], kh);
    v8f s0 = {}, s1 = {};
    s0 = WMMA_F32_F16(aq0, b00, s0);
    s1 = WMMA_F32_F16(aq0, b10, s1);
    s0 = WMMA_F32_F16(aq1, b01, s0);
    s1 = WMMA_F32_F16(aq1, b11, s1);

#pragma unroll
    for (int r = 0; r < 8; ++r) {                    // online softmax, row = r + 8*kh
      float a = s0[r] * scale, c = s1[r] * scale;
      float mx = rmax16(fmaxf(a, c));                // DPP row max (no LDS)
      float mnew = fmaxf(mrow[r], mx);
      float fac = __expf(mrow[r] - mnew);
      float p0 = __expf(a - mnew), p1 = __expf(c - mnew);
      mrow[r] = mnew;
      o0[r] *= fac; o1[r] *= fac; o2[r] *= fac; o3[r] *= fac;
      osum[r] *= fac;                                // row-sum accumulator, same rescale
      int prow = r + 8 * kh;                         // C-layout -> LDS
      lP[wave][prow][ln]      = (_Float16)p0;
      lP[wave][prow][16 + ln] = (_Float16)p1;
    }
    asm volatile("s_wait_dscnt 0x0" ::: "memory");   // same-wave DS order
    v16h ap  = load_a32k(&lP[wave][ln][0], kh);      // P as A operand (16x32)
    v16h bv0 = load_b32k(&lV[ 0 + ln][0], kh);
    v16h bv1 = load_b32k(&lV[16 + ln][0], kh);
    v16h bv2 = load_b32k(&lV[32 + ln][0], kh);
    v16h bv3 = load_b32k(&lV[48 + ln][0], kh);
    o0   = WMMA_F32_F16(ap, bv0, o0);
    o1   = WMMA_F32_F16(ap, bv1, o1);
    o2   = WMMA_F32_F16(ap, bv2, o2);
    o3   = WMMA_F32_F16(ap, bv3, o3);
    osum = WMMA_F32_F16(ap, ones, osum);             // row sums of P via matrix pipe
  }

#pragma unroll
  for (int r = 0; r < 8; ++r) {
    float inv = 1.0f / osum[r];
    size_t ro = ((size_t)b * T_ + t0 + r + 8 * kh) * E_ + hbase;
    O[ro +  0 + ln] = (_Float16)(o0[r] * inv);
    O[ro + 16 + ln] = (_Float16)(o1[r] * inv);
    O[ro + 32 + ln] = (_Float16)(o2[r] * inv);
    O[ro + 48 + ln] = (_Float16)(o3[r] * inv);
  }
}

extern "C" void kernel_launch(void* const* d_in, const int* in_sizes, int n_in,
                              void* d_out, int out_size, void* d_ws, size_t ws_size,
                              hipStream_t stream) {
  (void)in_sizes; (void)n_in; (void)out_size; (void)ws_size;
  const float* X  = (const float*)d_in[0];
  const int*   rp = (const int*)d_in[1];
  const float* Wq = (const float*)d_in[2];
  const float* bq = (const float*)d_in[3];
  const float* Wk = (const float*)d_in[4];
  const float* bk = (const float*)d_in[5];
  const float* Wv = (const float*)d_in[6];
  const float* bv = (const float*)d_in[7];
  const float* Wo = (const float*)d_in[8];
  const float* bo = (const float*)d_in[9];
  float* out = (float*)d_out;

  const size_t EE = (size_t)E_ * E_;   // 1M
  const size_t XN = (size_t)BT_ * E_;  // 16M
  _Float16* Wqh = (_Float16*)d_ws;
  _Float16* Wkh = Wqh + EE;
  _Float16* Wvh = Wkh + EE;
  _Float16* Woh = Wvh + EE;
  _Float16* Xh  = Woh + EE;            // reused as attention output
  _Float16* Qb  = Xh + XN;
  _Float16* Kb  = Qb + XN;
  _Float16* Vb  = Kb + XN;
  _Float16* Ob  = Xh;

  cvt_f32_f16<<<(int)((XN + 255) / 256), 256, 0, stream>>>(X,  Xh,  (int)XN);
  cvt_f32_f16<<<(int)((EE + 255) / 256), 256, 0, stream>>>(Wq, Wqh, (int)EE);
  cvt_f32_f16<<<(int)((EE + 255) / 256), 256, 0, stream>>>(Wk, Wkh, (int)EE);
  cvt_f32_f16<<<(int)((EE + 255) / 256), 256, 0, stream>>>(Wv, Wvh, (int)EE);
  cvt_f32_f16<<<(int)((EE + 255) / 256), 256, 0, stream>>>(Wo, Woh, (int)EE);

  dim3 gg((BT_ / 128) * (E_ / 128));   // 1024 WGs
  gemm_nt<false><<<gg, 256, 0, stream>>>(Xh, Wqh, bq, Qb, BT_, E_, E_);
  gemm_nt<false><<<gg, 256, 0, stream>>>(Xh, Wkh, bk, Kb, BT_, E_, E_);
  gemm_nt<false><<<gg, 256, 0, stream>>>(Xh, Wvh, bv, Vb, BT_, E_, E_);

  rope2d<<<(BT_ * (E_ / 2) + 255) / 256, 256, 0, stream>>>(Qb, Kb, rp);

  attn_flash<<<B_ * H_ * (T_ / 128), 256, 0, stream>>>(Qb, Kb, Vb, Ob);

  gemm_nt<true><<<gg, 256, 0, stream>>>(Ob, Woh, bo, out, BT_, E_, E_);
}
